// MultiHeadAttention_70257075028209
// MI455X (gfx1250) — matrix operand order repaired and checked by reading, not compiled, not run on hardware
//
#include <hip/hip_runtime.h>
#include <hip/hip_bf16.h>

typedef __attribute__((ext_vector_type(16))) __bf16 v16bf;
typedef __attribute__((ext_vector_type(8)))  __bf16 v8bf;
typedef __attribute__((ext_vector_type(4)))  __bf16 v4bf;
typedef __attribute__((ext_vector_type(8)))  float  v8f;
typedef __attribute__((ext_vector_type(4)))  float  v4f;

#define DEV __device__ __forceinline__

// Problem constants
constexpr int S_LEN = 2048;
constexpr int D_DIM = 1024;
constexpr int N_HEAD = 16;
constexpr int DK = 64;
constexpr int M_ROWS = 2 * S_LEN; // 4096

DEV v16bf combine16(v8bf lo, v8bf hi) {
    v16bf r;
#pragma unroll
    for (int i = 0; i < 8; i++) { r[i] = lo[i]; r[i + 8] = hi[i]; }
    return r;
}

// ---- CDNA5 async global->LDS copy (ASYNCcnt-tracked, no VGPR data path) ----
DEV void async_copy_b128(void* lds_dst, const void* gsrc) {
    unsigned lds_off = (unsigned)(size_t)lds_dst;  // LDS aperture: addr[31:0] = LDS offset
    asm volatile("global_load_async_to_lds_b128 %0, %1, off"
                 :: "v"(lds_off), "v"(gsrc)
                 : "memory");
}
DEV void async_wait0() {
    asm volatile("s_wait_asynccnt 0x0" ::: "memory");
}

// ---------------------------------------------------------------------------
// GEMM: Y = X @ W^T + bias
//   X: [4096, 1024] (f32 if !IN_BF16, else bf16)
//   W: [1024, 1024] f32, row-major (N, K)  -> B[k][n] = W[n][k]
//   SPLIT_OUT=true : write bf16 to head-split layout [B, H, S, 64]
//   SPLIT_OUT=false: write f32 to [4096, 1024] (+bias)  (final output)
// Block: 256 thr (8 waves), tile 128(M) x 64(N), K-step 32.
// ---------------------------------------------------------------------------
template <bool IN_BF16, bool SPLIT_OUT>
__global__ __launch_bounds__(256)
void gemm_wmma_kernel(const void* __restrict__ Aptr,
                      const float* __restrict__ W,
                      const float* __restrict__ bias,
                      void* __restrict__ Out)
{
    constexpr int LDA = 40; // 32 + 8 pad (80B rows, conflict-free b128 reads)
    __shared__ __bf16 As[128 * LDA];
    __shared__ __bf16 Bs[64 * LDA];

    const int tid  = threadIdx.x;
    const int wave = tid >> 5;
    const int lane = tid & 31;
    const int l16  = lane & 15;
    const int khalf = lane >> 4;

    const int blockM = blockIdx.y * 128;
    const int blockN = blockIdx.x * 64;
    const int wm = (wave >> 1) * 32; // 0,32,64,96
    const int wn = (wave & 1) * 32;  // 0,32

    v8f acc[2][2] = {};

    for (int k0 = 0; k0 < D_DIM; k0 += 32) {
        __syncthreads();
        // ---- stage A tile (128x32) into LDS as bf16 ----
        if constexpr (!IN_BF16) {
            const float* A = (const float*)Aptr;
            const int r = tid >> 3;          // 0..31
            const int c = (tid & 7) * 4;     // 0..28
#pragma unroll
            for (int rr = 0; rr < 128; rr += 32) {
                v4f d = *(const v4f*)(A + (size_t)(blockM + rr + r) * D_DIM + k0 + c);
                v4bf t = { (__bf16)d.x, (__bf16)d.y, (__bf16)d.z, (__bf16)d.w };
                *(v4bf*)(&As[(rr + r) * LDA + c]) = t;
            }
        } else {
            // raw bf16 copy: use async global->LDS (no VGPR staging)
            const __bf16* A = (const __bf16*)Aptr;
            const int r = tid >> 2;          // 0..63
            const int c = (tid & 3) * 8;     // 0,8,16,24
#pragma unroll
            for (int rr = 0; rr < 128; rr += 64) {
                async_copy_b128(&As[(rr + r) * LDA + c],
                                A + (size_t)(blockM + rr + r) * D_DIM + k0 + c);
            }
        }
        // ---- stage W tile (64 rows x 32 k) into LDS as bf16 (f32 convert) ----
        {
            const int r = tid >> 3;
            const int c = (tid & 7) * 4;
#pragma unroll
            for (int rr = 0; rr < 64; rr += 32) {
                v4f d = *(const v4f*)(W + (size_t)(blockN + rr + r) * D_DIM + k0 + c);
                v4bf t = { (__bf16)d.x, (__bf16)d.y, (__bf16)d.z, (__bf16)d.w };
                *(v4bf*)(&Bs[(rr + r) * LDA + c]) = t;
            }
        }
        if constexpr (IN_BF16) async_wait0();
        __syncthreads();

        // ---- fragments per ISA VGPR layouts ----
        v16bf aF[2], bF[2];
#pragma unroll
        for (int i = 0; i < 2; i++) {
            const __bf16* base = &As[(wm + i * 16 + l16) * LDA + khalf * 8];
            aF[i] = combine16(*(const v8bf*)base, *(const v8bf*)(base + 16));
        }
#pragma unroll
        for (int j = 0; j < 2; j++) {
            const __bf16* base = &Bs[(wn + j * 16 + l16) * LDA + khalf * 8];
            bF[j] = combine16(*(const v8bf*)base, *(const v8bf*)(base + 16));
        }
#pragma unroll
        for (int i = 0; i < 2; i++)
#pragma unroll
            for (int j = 0; j < 2; j++)
                acc[i][j] = __builtin_amdgcn_wmma_f32_16x16x32_bf16(
                    false, aF[i], false, bF[j], (short)0, acc[i][j], false, false);
    }

    // ---- epilogue ----
#pragma unroll
    for (int i = 0; i < 2; i++)
#pragma unroll
        for (int j = 0; j < 2; j++)
#pragma unroll
            for (int r = 0; r < 8; r++) {
                const int m = blockM + wm + i * 16 + r + khalf * 8;
                const int n = blockN + wn + j * 16 + l16;
                const float v = acc[i][j][r] + bias[n];
                if constexpr (SPLIT_OUT) {
                    const int b = m >> 11, s = m & 2047;
                    const int h = n >> 6, dk = n & 63;
                    ((__bf16*)Out)[(((size_t)(b * N_HEAD + h)) * S_LEN + s) * DK + dk] = (__bf16)v;
                } else {
                    ((float*)Out)[(size_t)m * D_DIM + n] = v;
                }
            }
}

// ---------------------------------------------------------------------------
// Flash attention: block = 128 query rows of one (b,h); 8 waves x 16 rows.
// KV tile = 64 keys (amortizes the cross-lane softmax reductions + barriers).
// Q/K tiles staged via async global->LDS; V staged transposed via VGPR path.
// ---------------------------------------------------------------------------
__global__ __launch_bounds__(256)
void flash_attn_kernel(const __bf16* __restrict__ Q,
                       const __bf16* __restrict__ K,
                       const __bf16* __restrict__ V,
                       __bf16* __restrict__ Ctx)
{
    constexpr int KV = 64;
    constexpr int LQ = 72, LK = 72, LV = 72, LP = 72;
    __shared__ __bf16 Qs[128 * LQ];
    __shared__ __bf16 Ks[KV * LK];
    __shared__ __bf16 Vt[64 * LV];      // transposed: Vt[dk][kv]
    __shared__ __bf16 Ps[8 * 16 * LP];  // per-wave P scratch (C-layout -> A-layout)

    const int tid  = threadIdx.x;
    const int wave = tid >> 5;
    const int lane = tid & 31;
    const int l16  = lane & 15;
    const int khalf = lane >> 4;

    const int qt = blockIdx.x;  // query tile (0..15)
    const int bh = blockIdx.y;  // 0..31
    const size_t base = (size_t)bh * S_LEN * DK;

    // ---- stage Q tile (128x64) via async copies ----
    {
        const int r = tid >> 1;
        const int c = (tid & 1) * 32;
#pragma unroll
        for (int cc = 0; cc < 32; cc += 8)
            async_copy_b128(&Qs[r * LQ + c + cc],
                            Q + base + (size_t)(qt * 128 + r) * DK + c + cc);
    }
    async_wait0();
    __syncthreads();

    v16bf aQ[2];
#pragma unroll
    for (int kk = 0; kk < 2; kk++) {
        const __bf16* b = &Qs[(wave * 16 + l16) * LQ + kk * 32 + khalf * 8];
        aQ[kk] = combine16(*(const v8bf*)b, *(const v8bf*)(b + 16));
    }

    v8f ctx[4] = {};
    float mrow[8], lrow[8];
#pragma unroll
    for (int r = 0; r < 8; r++) { mrow[r] = -1e30f; lrow[r] = 0.f; }

    for (int kt = 0; kt < S_LEN / KV; kt++) {
        __syncthreads();
        {   // K tile 64x64 via async copies (2 per thread)
            const int r = tid >> 2, c = (tid & 3) * 16;
            async_copy_b128(&Ks[r * LK + c],
                            K + base + (size_t)(kt * KV + r) * DK + c);
            async_copy_b128(&Ks[r * LK + c + 8],
                            K + base + (size_t)(kt * KV + r) * DK + c + 8);
        }
        {   // V tile 64x64, transposed into Vt[dk][kv]
            const int r = tid >> 2, c = (tid & 3) * 16;
#pragma unroll
            for (int cc = 0; cc < 16; cc += 8) {
                v8bf d = *(const v8bf*)(V + base + (size_t)(kt * KV + r) * DK + c + cc);
#pragma unroll
                for (int e = 0; e < 8; e++) Vt[(c + cc + e) * LV + r] = d[e];
            }
        }
        async_wait0();
        __syncthreads();

        // ---- scores S = Q K^T (16 q-rows x 64 keys): 4 n-tiles ----
        v8f sc[4] = {};
#pragma unroll
        for (int kk = 0; kk < 2; kk++)
#pragma unroll
            for (int nt = 0; nt < 4; nt++) {
                const __bf16* b = &Ks[(nt * 16 + l16) * LK + kk * 32 + khalf * 8];
                v16bf kf = combine16(*(const v8bf*)b, *(const v8bf*)(b + 16));
                sc[nt] = __builtin_amdgcn_wmma_f32_16x16x32_bf16(
                    false, aQ[kk], false, kf, (short)0, sc[nt], false, false);
            }

        // ---- online softmax update (row lives across 16 lanes) ----
#pragma unroll
        for (int r = 0; r < 8; r++) {
            float x[4];
#pragma unroll
            for (int nt = 0; nt < 4; nt++) x[nt] = sc[nt][r] * 0.125f; // 1/sqrt(64)
            float t = fmaxf(fmaxf(x[0], x[1]), fmaxf(x[2], x[3]));
#pragma unroll
            for (int m = 8; m >= 1; m >>= 1) t = fmaxf(t, __shfl_xor(t, m, 32));
            const float mnew = fmaxf(mrow[r], t);
            const float corr = __expf(mrow[r] - mnew);
            float p[4], sum = 0.f;
#pragma unroll
            for (int nt = 0; nt < 4; nt++) { p[nt] = __expf(x[nt] - mnew); sum += p[nt]; }
#pragma unroll
            for (int m = 8; m >= 1; m >>= 1) sum += __shfl_xor(sum, m, 32);
            lrow[r] = lrow[r] * corr + sum;
            mrow[r] = mnew;
#pragma unroll
            for (int j = 0; j < 4; j++) ctx[j][r] *= corr;
            const int prow = wave * 16 + r + khalf * 8;
#pragma unroll
            for (int nt = 0; nt < 4; nt++)
                Ps[prow * LP + nt * 16 + l16] = (__bf16)p[nt];
        }

        // ---- re-read P as 16x64 A fragments, then ctx += P @ V ----
        v16bf aP[2];
#pragma unroll
        for (int kk = 0; kk < 2; kk++) {
            const __bf16* pb = &Ps[(wave * 16 + l16) * LP + kk * 32 + khalf * 8];
            aP[kk] = combine16(*(const v8bf*)pb, *(const v8bf*)(pb + 16));
        }
#pragma unroll
        for (int kk = 0; kk < 2; kk++)
#pragma unroll
            for (int j = 0; j < 4; j++) {
                const __bf16* vb = &Vt[(j * 16 + l16) * LV + kk * 32 + khalf * 8];
                v16bf bV = combine16(*(const v8bf*)vb, *(const v8bf*)(vb + 16));
                ctx[j] = __builtin_amdgcn_wmma_f32_16x16x32_bf16(
                    false, aP[kk], false, bV, (short)0, ctx[j], false, false);
            }
    }

    // ---- epilogue: ctx /= l, write bf16 merged-head [b*s][h*64+dk] ----
    const int b = bh >> 4, h = bh & 15;
#pragma unroll
    for (int j = 0; j < 4; j++)
#pragma unroll
        for (int r = 0; r < 8; r++) {
            const int srow = qt * 128 + wave * 16 + r + khalf * 8;
            const float v = ctx[j][r] / lrow[r];
            Ctx[((size_t)(b * S_LEN + srow)) * D_DIM + h * 64 + j * 16 + l16] = (__bf16)v;
        }
}

// ---------------------------------------------------------------------------
extern "C" void kernel_launch(void* const* d_in, const int* in_sizes, int n_in,
                              void* d_out, int out_size, void* d_ws, size_t ws_size,
                              hipStream_t stream)
{
    (void)in_sizes; (void)n_in; (void)out_size; (void)ws_size;
    const float* query = (const float*)d_in[0];
    const float* key_  = (const float*)d_in[1];
    const float* value = (const float*)d_in[2];
    const float* Wq = (const float*)d_in[3];
    const float* bq = (const float*)d_in[4];
    const float* Wk = (const float*)d_in[5];
    const float* bk = (const float*)d_in[6];
    const float* Wv = (const float*)d_in[7];
    const float* bv = (const float*)d_in[8];
    const float* Wo = (const float*)d_in[9];
    const float* bo = (const float*)d_in[10];

    const size_t nElem = (size_t)M_ROWS * D_DIM;       // 4M elems
    char* ws = (char*)d_ws;
    __bf16* Qbf = (__bf16*)(ws + 0 * nElem * 2);
    __bf16* Kbf = (__bf16*)(ws + 1 * nElem * 2);
    __bf16* Vbf = (__bf16*)(ws + 2 * nElem * 2);
    __bf16* Cbf = (__bf16*)(ws + 3 * nElem * 2);

    dim3 blk(256);
    dim3 gGemm(D_DIM / 64, M_ROWS / 128);  // 16 x 32
    gemm_wmma_kernel<false, true><<<gGemm, blk, 0, stream>>>(query, Wq, bq, Qbf);
    gemm_wmma_kernel<false, true><<<gGemm, blk, 0, stream>>>(key_,  Wk, bk, Kbf);
    gemm_wmma_kernel<false, true><<<gGemm, blk, 0, stream>>>(value, Wv, bv, Vbf);

    dim3 gAttn(S_LEN / 128, 2 * N_HEAD);   // 16 x 32
    flash_attn_kernel<<<gAttn, blk, 0, stream>>>(Qbf, Kbf, Vbf, Cbf);

    gemm_wmma_kernel<true, false><<<gGemm, blk, 0, stream>>>(Cbf, Wo, bo, (float*)d_out);
}
